// LambdaConv1d_63788854280614
// MI455X (gfx1250) — matrix operand order repaired and checked by reading, not compiled, not run on hardware
//
#include <hip/hip_runtime.h>

#define EPS 1e-5f

typedef __attribute__((ext_vector_type(16))) __bf16        bf16x16;
typedef __attribute__((ext_vector_type(8)))  float         f32x8;
typedef __attribute__((ext_vector_type(4)))  unsigned int  u32x4;

union Frag {
    bf16x16 v;
    __bf16  h[16];
    u32x4   q[2];
};

__device__ __forceinline__ __bf16 f2bf(float f) { return (__bf16)f; }

// ---------------------------------------------------------------------------
// K1: pack [Wq;Wk;Wv] -> bf16 (384x128), emb -> bf16 padded (16 x 32), K=u*8+m
// ---------------------------------------------------------------------------
__global__ void prep_kernel(const float* __restrict__ Wq, const float* __restrict__ Wk,
                            const float* __restrict__ Wv, const float* __restrict__ emb,
                            __bf16* __restrict__ Wcat, __bf16* __restrict__ embp) {
    int idx = blockIdx.x * 256 + threadIdx.x;
    if (idx < 384 * 128) {
        int row = idx >> 7, col = idx & 127;
        float v;
        if (row < 64)       v = Wq[row * 128 + col];
        else if (row < 128) v = Wk[(row - 64) * 128 + col];
        else                v = Wv[(row - 128) * 128 + col];
        Wcat[idx] = f2bf(v);
    }
    if (idx < 512) {
        int k = idx >> 5, rr = idx & 31, u = rr >> 3, m = rr & 7;
        float v = (m < 7) ? emb[(k * 4 + u) * 7 + m] : 0.f;
        embp[idx] = f2bf(v);
    }
}

// ---------------------------------------------------------------------------
// K2: x (b,c,n) fp32 -> xT (b,n,c) bf16, 32x32 LDS tiles for coalescing
// ---------------------------------------------------------------------------
__global__ void transpose_x(const float* __restrict__ x, __bf16* __restrict__ xT) {
    __shared__ float tile[32][33];
    int tx = threadIdx.x & 31, ty = threadIdx.x >> 5;
    int nt = blockIdx.x & 127;
    int ct = (blockIdx.x >> 7) & 3;
    int b  = blockIdx.x >> 9;
    int n0 = nt * 32, c0 = ct * 32;
#pragma unroll
    for (int r = 0; r < 4; ++r) {
        int c = c0 + ty + r * 8;
        tile[ty + r * 8][tx] = x[((size_t)b * 128 + c) * 4096 + n0 + tx];
    }
    __syncthreads();
#pragma unroll
    for (int r = 0; r < 4; ++r) {
        int n = n0 + ty + r * 8;
        xT[((size_t)b * 4096 + n) * 128 + c0 + tx] = f2bf(tile[tx][ty + r * 8]);
    }
}

// ---------------------------------------------------------------------------
// K3: projection GEMM  P[b] (384x4096) = Wcat (384x128) @ x_b (128x4096)
// one wave = 16x64 output strip (4 C tiles); K loop 4 x 32, software-pipelined
// with double-buffered fragments so loads overlap WMMA issue.
// ---------------------------------------------------------------------------
__device__ __forceinline__ void load_fragA(Frag& f, const __bf16* base) {
    f.q[0] = *(const u32x4*)base;        // K = +0..7   (lane-half selected)
    f.q[1] = *(const u32x4*)(base + 16); // K = +16..23
}
__device__ __forceinline__ void load_fragB(Frag& f, const __bf16* base) {
    f.q[0] = *(const u32x4*)base;        // K = +0..7
    f.q[1] = *(const u32x4*)(base + 16); // K = +16..23
}

__global__ void gemm_proj(const __bf16* __restrict__ Wcat,
                          const __bf16* __restrict__ xT,
                          float* __restrict__ P) {
    int ng = blockIdx.x & 7;
    int mt = (blockIdx.x >> 3) % 24;
    int b  = blockIdx.x / (8 * 24);
    int lane = threadIdx.x & 31, w = threadIdx.x >> 5;
    int r = lane & 15, h = lane >> 4;
    int m0 = mt * 16;
    int nbase = ng * 512 + w * 64;
    const __bf16* xb = xT + (size_t)b * 4096 * 128;

    const __bf16* paBase = Wcat + (m0 + r) * 128 + h * 8;
    const __bf16* pbBase[4];
#pragma unroll
    for (int nt = 0; nt < 4; ++nt)
        pbBase[nt] = xb + (size_t)(nbase + nt * 16 + r) * 128 + h * 8;

    f32x8 acc[4] = {};
    Frag fa[2], fb[2][4];
    load_fragA(fa[0], paBase);
#pragma unroll
    for (int nt = 0; nt < 4; ++nt) load_fragB(fb[0][nt], pbBase[nt]);

#pragma unroll
    for (int ks = 0; ks < 4; ++ks) {
        int cur = ks & 1, nxt = cur ^ 1;
        if (ks < 3) {
            int k1 = (ks + 1) * 32;
            load_fragA(fa[nxt], paBase + k1);
#pragma unroll
            for (int nt = 0; nt < 4; ++nt) load_fragB(fb[nxt][nt], pbBase[nt] + k1);
        }
#pragma unroll
        for (int nt = 0; nt < 4; ++nt)
            acc[nt] = __builtin_amdgcn_wmma_f32_16x16x32_bf16(
                false, fa[cur].v, false, fb[cur][nt].v, (short)0, acc[nt], false, false);
    }

#pragma unroll
    for (int nt = 0; nt < 4; ++nt) {
        int n = nbase + nt * 16 + r;
#pragma unroll
        for (int i = 0; i < 8; ++i)
            P[((size_t)b * 384 + m0 + h * 8 + i) * 4096 + n] = acc[nt][i];
    }
}

// ---------------------------------------------------------------------------
// K4a: BN stats for q (P ch 0..63) and v (P ch 128..383) -> scale/bias
// ---------------------------------------------------------------------------
__global__ void bn_stats_kernel(const float* __restrict__ P,
                                const float* __restrict__ gq, const float* __restrict__ bq,
                                const float* __restrict__ gv, const float* __restrict__ bv,
                                float* __restrict__ sc, float* __restrict__ bs) {
    __shared__ float s1[256], s2[256];
    int ch = blockIdx.x;                       // 0..319
    int pc = (ch < 64) ? ch : 64 + ch;         // P channel
    float g  = (ch < 64) ? gq[ch] : gv[ch - 64];
    float be = (ch < 64) ? bq[ch] : bv[ch - 64];
    float a = 0.f, a2 = 0.f;
    for (int idx = threadIdx.x; idx < 32 * 4096; idx += 256) {
        int b = idx >> 12, n = idx & 4095;
        float v = P[((size_t)b * 384 + pc) * 4096 + n];
        a += v; a2 += v * v;
    }
    s1[threadIdx.x] = a; s2[threadIdx.x] = a2;
    __syncthreads();
    for (int off = 128; off > 0; off >>= 1) {
        if ((int)threadIdx.x < off) {
            s1[threadIdx.x] += s1[threadIdx.x + off];
            s2[threadIdx.x] += s2[threadIdx.x + off];
        }
        __syncthreads();
    }
    if (threadIdx.x == 0) {
        float inv   = 1.f / (32.f * 4096.f);
        float mean  = s1[0] * inv;
        float var   = s2[0] * inv - mean * mean;
        float scale = g * rsqrtf(var + EPS);
        sc[pc] = scale;
        bs[pc] = be - mean * scale;
    }
}

// ---------------------------------------------------------------------------
// K4b: softmax max & 1/sum(exp) per (b, k-channel); P ch 64..127
// ---------------------------------------------------------------------------
__global__ void softmax_stats_kernel(const float* __restrict__ P,
                                     float* __restrict__ kmax, float* __restrict__ kinv) {
    __shared__ float sm[256];
    int ch = blockIdx.x & 63, b = blockIdx.x >> 6;
    const float* row = P + ((size_t)b * 384 + 64 + ch) * 4096;
    float m = -3.4e38f;
    for (int n = threadIdx.x; n < 4096; n += 256) m = fmaxf(m, row[n]);
    sm[threadIdx.x] = m; __syncthreads();
    for (int off = 128; off > 0; off >>= 1) {
        if ((int)threadIdx.x < off) sm[threadIdx.x] = fmaxf(sm[threadIdx.x], sm[threadIdx.x + off]);
        __syncthreads();
    }
    float mx = sm[0];
    __syncthreads();
    float s = 0.f;
    for (int n = threadIdx.x; n < 4096; n += 256) s += __expf(row[n] - mx);
    sm[threadIdx.x] = s; __syncthreads();
    for (int off = 128; off > 0; off >>= 1) {
        if ((int)threadIdx.x < off) sm[threadIdx.x] += sm[threadIdx.x + off];
        __syncthreads();
    }
    if (threadIdx.x == 0) { kmax[b * 64 + ch] = mx; kinv[b * 64 + ch] = 1.f / sm[0]; }
}

// ---------------------------------------------------------------------------
// K5: lam_c[b,k,v] = sum_{u,n} softmax(k)[b,k,u,n] * BN(v)[b,v,u,n]
// WMMA with K = (u,n) = 16384; one wave per (b, v-tile)
// ---------------------------------------------------------------------------
__global__ void lamc_kernel(const float* __restrict__ P,
                            const float* __restrict__ sc, const float* __restrict__ bs,
                            const float* __restrict__ kmax, const float* __restrict__ kinv,
                            float* __restrict__ lamc) {
    int b = blockIdx.x;
    int lane = threadIdx.x & 31;
    int w = threadIdx.x >> 5;             // v-tile 0..3
    int c16 = lane & 15, h = lane >> 4;
    int v = w * 16 + c16;                 // B column (v index)

    f32x8 acc = {};
    for (int u = 0; u < 4; ++u) {
        int kch = c16 * 4 + u;            // A row's k-channel (within k block)
        const float* rowA = P + ((size_t)b * 384 + 64 + kch) * 4096;
        float mx  = kmax[b * 64 + kch];
        float inv = kinv[b * 64 + kch];
        int vch = 128 + v * 4 + u;
        const float* rowB = P + ((size_t)b * 384 + vch) * 4096;
        float scb = sc[vch], bsb = bs[vch];

        for (int n0 = 0; n0 < 4096; n0 += 32) {
            Frag fa, fb;
#pragma unroll
            for (int g = 0; g < 2; ++g) {
                const float* base = rowA + n0 + h * 16 + g * 8;
#pragma unroll
                for (int j = 0; j < 8; ++j)
                    fa.h[g * 8 + j] = f2bf(__expf(base[j] - mx) * inv);
            }
#pragma unroll
            for (int j = 0; j < 8; ++j) {
                const float* p = rowB + n0 + h * 16 + 2 * j;
                fb.h[2 * j]     = f2bf(p[0] * scb + bsb);
                fb.h[2 * j + 1] = f2bf(p[1] * scb + bsb);
            }
            acc = __builtin_amdgcn_wmma_f32_16x16x32_bf16(
                false, fa.v, false, fb.v, (short)0, acc, false, false);
        }
    }
#pragma unroll
    for (int i = 0; i < 8; ++i)
        lamc[(size_t)b * 1024 + (h * 8 + i) * 64 + v] = acc[i];
}

// ---------------------------------------------------------------------------
// K6: per (b, 16-wide n-tile): lam_p via WMMA (A = emb 16x32, K=u*8+m),
// fold lam_c, then y[h,v,n] = sum_k q*Lam with cross-half shfl reduction.
// lam_c tile staged via CDNA5 async DMA (global_load_async_to_lds_b128).
// ---------------------------------------------------------------------------
__global__ void final_kernel(const float* __restrict__ P,
                             const float* __restrict__ sc, const float* __restrict__ bs,
                             const float* __restrict__ lamc,
                             const __bf16* __restrict__ embp,
                             float* __restrict__ out) {
    __shared__ float vs[256 * 24];   // BN'd v, window n0-3 .. n0+18 (cols 0..21)
    __shared__ float qs[1024];       // BN'd q: [h*256 + k*16 + nn]
    __shared__ float lc[1024];       // lam_c[b]: [k*64 + v]

    int b = blockIdx.x >> 8, nt = blockIdx.x & 255;
    int n0 = nt * 16;
    int tid = threadIdx.x;

    // lam_c: pure copy -> async DMA global->LDS, 256 lanes x 16B = 4KB.
    {
        unsigned ldsa = (unsigned)(size_t)(&lc[tid * 4]);   // LDS byte address (low 32b of flat)
        unsigned voff = (unsigned)tid * 16u;
        const float* gsrc = lamc + (size_t)b * 1024;
        asm volatile("global_load_async_to_lds_b128 %0, %1, %2"
                     :: "v"(ldsa), "v"(voff), "s"(gsrc)
                     : "memory");
    }

    for (int i = tid; i < 1024; i += 256) {
        int ch = i >> 4, nn = i & 15;             // ch = h*16+k
        qs[i] = P[((size_t)b * 384 + ch) * 4096 + n0 + nn] * sc[ch] + bs[ch];
    }
    for (int i = tid; i < 6144; i += 256) {
        int ch = i / 24, j = i % 24;
        int n = n0 - 3 + j;
        float val = 0.f;
        if (j < 22 && n >= 0 && n < 4096)
            val = P[((size_t)b * 384 + 128 + ch) * 4096 + n] * sc[128 + ch] + bs[128 + ch];
        vs[i] = val;
    }
    asm volatile("s_wait_asynccnt 0x0" ::: "memory");
    __syncthreads();

    int lane = tid & 31, w = tid >> 5;
    int nn = lane & 15, h = lane >> 4;

    float qreg[4][8];
#pragma unroll
    for (int hh = 0; hh < 4; ++hh)
#pragma unroll
        for (int i = 0; i < 8; ++i)
            qreg[hh][i] = qs[hh * 256 + (h * 8 + i) * 16 + nn];

    Frag fa;   // A = emb_pad (16 x 32)
    fa.q[0] = *(const u32x4*)(embp + (lane & 15) * 32 + h * 8);
    fa.q[1] = *(const u32x4*)(embp + (lane & 15) * 32 + 16 + h * 8);

    for (int jj = 0; jj < 8; ++jj) {
        int t = w * 8 + jj;                       // v index 0..63
        Frag fb;
#pragma unroll
        for (int j = 0; j < 8; ++j) {
            int K0 = (j >> 2) * 16 + h * 8 + 2 * (j & 3);
            int u = K0 >> 3, m0 = K0 & 7;         // m0 in {0,2,4,6}
            const float* p = &vs[(t * 4 + u) * 24 + nn + m0];
            fb.h[2 * j]     = f2bf(p[0]);
            fb.h[2 * j + 1] = f2bf((m0 == 6) ? 0.f : p[1]);
        }
        f32x8 c = {};
        c = __builtin_amdgcn_wmma_f32_16x16x32_bf16(
            false, fa.v, false, fb.v, (short)0, c, false, false);

        float y0 = 0.f, y1 = 0.f, y2 = 0.f, y3 = 0.f;
#pragma unroll
        for (int i = 0; i < 8; ++i) {
            float lam = c[i] + lc[(h * 8 + i) * 64 + t];  // lam_p + lam_c
            y0 += qreg[0][i] * lam;
            y1 += qreg[1][i] * lam;
            y2 += qreg[2][i] * lam;
            y3 += qreg[3][i] * lam;
        }
        // sum the two k-halves (rows 0..7 vs rows 8..15 live in opposite lane halves)
        y0 += __shfl_xor(y0, 16);
        y1 += __shfl_xor(y1, 16);
        y2 += __shfl_xor(y2, 16);
        y3 += __shfl_xor(y3, 16);
        if (lane < 16) {
            size_t base = (size_t)b * 256 * 4096 + (size_t)n0 + nn;
            out[base + (size_t)(0 * 64 + t) * 4096] = y0;
            out[base + (size_t)(1 * 64 + t) * 4096] = y1;
            out[base + (size_t)(2 * 64 + t) * 4096] = y2;
            out[base + (size_t)(3 * 64 + t) * 4096] = y3;
        }
    }
}

// ---------------------------------------------------------------------------
extern "C" void kernel_launch(void* const* d_in, const int* in_sizes, int n_in,
                              void* d_out, int out_size, void* d_ws, size_t ws_size,
                              hipStream_t stream) {
    const float* x   = (const float*)d_in[0];
    const float* Wq  = (const float*)d_in[1];
    const float* gq  = (const float*)d_in[2];
    const float* bq  = (const float*)d_in[3];
    const float* Wk  = (const float*)d_in[4];
    const float* Wv  = (const float*)d_in[5];
    const float* gv  = (const float*)d_in[6];
    const float* bv  = (const float*)d_in[7];
    const float* emb = (const float*)d_in[8];
    float* out = (float*)d_out;

    char* ws = (char*)d_ws;
    size_t off = 0;
    float*  P    = (float*)(ws + off);  off += (size_t)32 * 384 * 4096 * 4; // 192 MB
    __bf16* xT   = (__bf16*)(ws + off); off += (size_t)32 * 4096 * 128 * 2; // 32 MB
    __bf16* Wcat = (__bf16*)(ws + off); off += 384 * 128 * 2;
    __bf16* embp = (__bf16*)(ws + off); off += 1024;
    float* sc    = (float*)(ws + off);  off += 1536;
    float* bsA   = (float*)(ws + off);  off += 1536;
    float* kmax  = (float*)(ws + off);  off += 8192;
    float* kinv  = (float*)(ws + off);  off += 8192;
    float* lamc  = (float*)(ws + off);  off += 131072;
    (void)ws_size; (void)in_sizes; (void)n_in; (void)out_size;

    prep_kernel<<<192, 256, 0, stream>>>(Wq, Wk, Wv, emb, Wcat, embp);
    transpose_x<<<16384, 256, 0, stream>>>(x, xT);
    gemm_proj<<<6144, 256, 0, stream>>>(Wcat, xT, P);
    bn_stats_kernel<<<320, 256, 0, stream>>>(P, gq, bq, gv, bv, sc, bsA);
    softmax_stats_kernel<<<2048, 256, 0, stream>>>(P, kmax, kinv);
    lamc_kernel<<<32, 128, 0, stream>>>(P, sc, bsA, kmax, kinv, lamc);
    final_kernel<<<8192, 256, 0, stream>>>(P, sc, bsA, lamc, embp, out);
}
